// MultiHeadedAttention_38130719653892
// MI455X (gfx1250) — matrix operand order repaired and checked by reading, not compiled, not run on hardware
//
#include <hip/hip_runtime.h>

// ---------------------------------------------------------------------------
// Fused temporal multi-head attention for MI455X (gfx1250, wave32, WMMA).
// One workgroup per (batch b, head h). 352 threads = 11 wave32s; each wave
// owns one 16-row strip of the 176-long sequence. All intermediates (x, W
// slices, Q, K, V, scores) live in LDS; only x is read and out written to HBM.
// All GEMMs use v_wmma_f32_16x16x32_bf16.
// ---------------------------------------------------------------------------

typedef __attribute__((ext_vector_type(16))) __bf16 v16bf;
typedef __attribute__((ext_vector_type(8)))  float  v8f;

union BFrag {
    v16bf  v;
    uint4  q[2];
    __bf16 h[16];
};

__device__ __forceinline__ __bf16 f2bf(float f) { return (__bf16)f; }

// ---- problem constants -----------------------------------------------------
#define TLEN   8
#define JOINT  22
#define SEQ    176            // 8*22 = 11*16
#define HN     8
#define HD     32
#define DIN    128
#define DMODEL 256
#define NB     256
#define NTHREADS 352          // 11 wave32s

// ---- LDS layout (bytes), all regions 16B aligned ---------------------------
#define XST  136              // x strip stride (bf16): 128 + 8 pad  -> bank-safe
#define WST  136              // wT stride (bf16): 128 + 8 pad
#define QST  40               // Q/K stride (bf16): 32 + 8 pad
#define VTST 200              // vT stride (bf16): 176 rounded up, zero-padded tail
#define SST  196              // score stride (f32): 176 + pad, 16B-aligned rows

#define Q_OFF  0
#define K_OFF  (Q_OFF + SEQ*QST*2)                 // 14080
#define VT_OFF (K_OFF + SEQ*QST*2)                 // 28160
#define PH_OFF (VT_OFF + HD*VTST*2)                // 40960 (phase-overlapped)
#define XS_OFF PH_OFF                              // phase 1: x tile
#define WT_OFF (XS_OFF + SEQ*XST*2)                // 88832: 3 transposed W slices
#define SC_OFF PH_OFF                              // phase 2: scores (f32)
#define SMEM_BYTES (SC_OFF + SEQ*SST*4)            // 178944 < 320 KB

__global__ __launch_bounds__(NTHREADS, 1)
void attn_fused_gfx1250(const float* __restrict__ x,
                        const float* __restrict__ Wq, const float* __restrict__ bq,
                        const float* __restrict__ Wk, const float* __restrict__ bk,
                        const float* __restrict__ Wv, const float* __restrict__ bv,
                        float* __restrict__ out)
{
    extern __shared__ char smem[];
    __bf16* Qs  = (__bf16*)(smem + Q_OFF);
    __bf16* Ks  = (__bf16*)(smem + K_OFF);
    __bf16* VTs = (__bf16*)(smem + VT_OFF);   // V transposed: [d][t]
    __bf16* XS  = (__bf16*)(smem + XS_OFF);
    __bf16* WT  = (__bf16*)(smem + WT_OFF);   // [mat][n][k] (W transposed slice)
    float*  SC  = (float*)(smem + SC_OFF);

    const int h    = blockIdx.x;              // head
    const int b    = blockIdx.y;              // batch
    const int tid  = threadIdx.x;
    const int lane = tid & 31;
    const int wid  = tid >> 5;                // wave id == row strip (0..10)
    const int ln16 = lane & 15;
    const int kh   = lane >> 4;               // lane half (K-split per ISA tables)

    // ======== Phase 0: stage x[b] (f32->bf16) and W head-slices into LDS =====
    const float* xb = x + (size_t)b * SEQ * DIN;
    for (int i = tid; i < SEQ * DIN / 4; i += NTHREADS) {
        int row = i >> 5, c4 = i & 31;        // 32 float4 per row
        float4 f = ((const float4*)xb)[i];
        union { __bf16 hh[4]; uint2 u; } cv;
        cv.hh[0] = f2bf(f.x); cv.hh[1] = f2bf(f.y);
        cv.hh[2] = f2bf(f.z); cv.hh[3] = f2bf(f.w);
        *(uint2*)(XS + row * XST + c4 * 4) = cv.u;
    }
    // W slices, transposed: WT[mat][n][k] = W[k][h*32+n]  (lanes sweep n -> coalesced)
    for (int i = tid; i < 3 * HD * DIN; i += NTHREADS) {
        int mat = i >> 12;                    // 4096 elements per matrix
        int r   = i & 4095;
        int k   = r >> 5;
        int n   = r & 31;
        const float* wm = (mat == 0) ? Wq : (mat == 1) ? Wk : Wv;
        WT[(mat * HD + n) * WST + k] = f2bf(wm[k * DMODEL + h * HD + n]);
    }
    // zero vT (tail cols 176..199 must stay zero for the padded AV reduction)
    for (int i = tid; i < HD * VTST / 2; i += NTHREADS)
        ((unsigned int*)VTs)[i] = 0u;
    __syncthreads();

    // ======== Phase 1: QKV projection (per-strip), K-dim = 128 = 4 WMMA steps
    BFrag xa[4];
    {
        const __bf16* base = XS + (wid * 16 + ln16) * XST;
#pragma unroll
        for (int k4 = 0; k4 < 4; ++k4) {      // A 16x32 bf16 layout per ISA
            int ko = k4 * 32 + kh * 8;
            xa[k4].q[0] = *(const uint4*)(base + ko);        // K j..j+7
            xa[k4].q[1] = *(const uint4*)(base + ko + 16);   // K j+16..j+23
        }
    }
#pragma unroll
    for (int mat = 0; mat < 3; ++mat) {
        const float* bp = (mat == 0) ? bq : (mat == 1) ? bk : bv;
#pragma unroll
        for (int nt = 0; nt < 2; ++nt) {
            float bias = bp[h * HD + nt * 16 + ln16];
            v8f acc = {};
#pragma unroll
            for (int k4 = 0; k4 < 4; ++k4) {
                BFrag bf;                     // B 32x16: lane n = column n
                const __bf16* wb = WT + (mat * HD + nt * 16 + ln16) * WST
                                      + k4 * 32 + kh * 8;
                bf.q[0] = *(const uint4*)(wb);
                bf.q[1] = *(const uint4*)(wb + 16);
                acc = __builtin_amdgcn_wmma_f32_16x16x32_bf16(
                          false, xa[k4].v, false, bf.v, (short)0, acc, false, false);
            }
#pragma unroll
            for (int r = 0; r < 8; ++r) {     // C/D: VGPR r <-> M = r + 8*half
                float val = acc[r] + bias;
                int s = wid * 16 + r + kh * 8;
                int d = nt * 16 + ln16;
                if (mat == 0)      Qs[s * QST + d] = f2bf(val);
                else if (mat == 1) Ks[s * QST + d] = f2bf(val);
                else               VTs[d * VTST + s] = f2bf(val > 0.f ? val : 0.f);
            }
        }
    }
    __syncthreads();   // only barrier: K/V from all strips needed below

    // ======== Phase 2: scores strip = Q_strip @ K^T  (K-dim = HD = 32) =======
    BFrag qa;
    {
        const __bf16* base = Qs + (wid * 16 + ln16) * QST + kh * 8;
        qa.q[0] = *(const uint4*)(base);
        qa.q[1] = *(const uint4*)(base + 16);
    }
#pragma unroll
    for (int tt = 0; tt < 11; ++tt) {
        BFrag kb;                              // B col n = K row (t), d in table order
        const __bf16* kp = Ks + (tt * 16 + ln16) * QST + kh * 8;
        kb.q[0] = *(const uint4*)(kp);
        kb.q[1] = *(const uint4*)(kp + 16);
        v8f acc = {};
        acc = __builtin_amdgcn_wmma_f32_16x16x32_bf16(
                  false, qa.v, false, kb.v, (short)0, acc, false, false);
#pragma unroll
        for (int r = 0; r < 8; ++r)
            SC[(wid * 16 + r + kh * 8) * SST + tt * 16 + ln16] = acc[r];
    }

    // ======== Phase 3: masked softmax (strip-local, mask folded as predicate)
    {
        const float rscale = 0.17677669529663687f;   // 1/sqrt(32)
        int s  = wid * 16 + ln16;                    // lane pair (kh) splits row
        int fs = s / JOINT;
        int j0 = kh * 88;
        float mx = -3.4e38f;
        for (int j = j0; j < j0 + 88; ++j) {
            bool valid = (j == s) || (j / JOINT != fs);
            if (valid) mx = fmaxf(mx, SC[s * SST + j] * rscale);
        }
        mx = fmaxf(mx, __shfl_xor(mx, 16, 32));
        float sum = 0.f;
        for (int j = j0; j < j0 + 88; ++j) {
            bool valid = (j == s) || (j / JOINT != fs);
            float e = valid ? __expf(SC[s * SST + j] * rscale - mx) : 0.f;
            SC[s * SST + j] = e;
            sum += e;
        }
        sum += __shfl_xor(sum, 16, 32);
        float inv = 1.f / sum;
        for (int j = j0; j < j0 + 88; ++j) SC[s * SST + j] *= inv;
        for (int j = 176 + kh * 8; j < 184 + kh * 8; ++j) SC[s * SST + j] = 0.f; // pad
    }

    // ======== Phase 4: out strip = att_strip @ V  (K = 176 -> 6x32 padded) ==
#pragma unroll
    for (int dt = 0; dt < 2; ++dt) {
        v8f acc = {};
#pragma unroll
        for (int kt = 0; kt < 6; ++kt) {
            int t0 = kt * 32;
            BFrag af;                          // att f32 -> bf16 on the fly
            const float* ap = SC + (wid * 16 + ln16) * SST + t0 + kh * 8;
#pragma unroll
            for (int j = 0; j < 8; ++j) af.h[j]     = f2bf(ap[j]);
#pragma unroll
            for (int j = 0; j < 8; ++j) af.h[8 + j] = f2bf(ap[16 + j]);
            BFrag vb;                          // B col n = output dim d
            const __bf16* vp = VTs + (dt * 16 + ln16) * VTST + t0 + kh * 8;
            vb.q[0] = *(const uint4*)(vp);
            vb.q[1] = *(const uint4*)(vp + 16);
            acc = __builtin_amdgcn_wmma_f32_16x16x32_bf16(
                      false, af.v, false, vb.v, (short)0, acc, false, false);
        }
#pragma unroll
        for (int r = 0; r < 8; ++r) {
            int s = wid * 16 + r + kh * 8;
            int d = dt * 16 + ln16;
            out[((size_t)b * SEQ + s) * DMODEL + h * HD + d] = acc[r];
        }
    }
}

extern "C" void kernel_launch(void* const* d_in, const int* in_sizes, int n_in,
                              void* d_out, int out_size, void* d_ws, size_t ws_size,
                              hipStream_t stream) {
    (void)in_sizes; (void)n_in; (void)out_size; (void)d_ws; (void)ws_size;
    const float* x  = (const float*)d_in[0];
    const float* Wq = (const float*)d_in[1];
    const float* bq = (const float*)d_in[2];
    const float* Wk = (const float*)d_in[3];
    const float* bk = (const float*)d_in[4];
    const float* Wv = (const float*)d_in[5];
    const float* bv = (const float*)d_in[6];
    float* out = (float*)d_out;

    // Dynamic LDS > 64 KB: raise the limit (idempotent, graph-capture safe).
    hipFuncSetAttribute((const void*)attn_fused_gfx1250,
                        hipFuncAttributeMaxDynamicSharedMemorySize, SMEM_BYTES);

    dim3 grid(HN, NB);   // one workgroup per (head, batch)
    attn_fused_gfx1250<<<grid, NTHREADS, SMEM_BYTES, stream>>>(
        x, Wq, bq, Wk, bk, Wv, bv, out);
}
